// MultiHeadAttention_35613868818473
// MI455X (gfx1250) — matrix operand order repaired and checked by reading, not compiled, not run on hardware
//
#include <hip/hip_runtime.h>

#define HIDDEN 2048
#define HEADS 16
#define HEAD_DIM 128
#define BATCH 2
#define SEQ 2048

typedef __attribute__((ext_vector_type(16))) __bf16 v16bf;
typedef __attribute__((ext_vector_type(8)))  float  v8f;

// 16-byte POD vector types (trivially copyable for __builtin_bit_cast)
struct alignas(16) U4 { unsigned int x, y, z, w; };
struct alignas(16) F4 { float x, y, z, w; };
struct U32x8 { U4 lo, hi; };  // 32 bytes == v16bf

__device__ __forceinline__ unsigned short f2bf(float f) {
    unsigned int u = __builtin_bit_cast(unsigned int, f);
    u += 0x7FFFu + ((u >> 16) & 1u);   // round-to-nearest-even
    return (unsigned short)(u >> 16);
}

// Build a 16x bf16 A/B fragment from two 16-byte chunks.
__device__ __forceinline__ v16bf frag2(const unsigned short* p0, const unsigned short* p1) {
    U32x8 r;
    r.lo = *(const U4*)p0;
    r.hi = *(const U4*)p1;
    return __builtin_bit_cast(v16bf, r);
}

// gfx1250 async global->LDS copy (ASYNCcnt-tracked, bypasses VGPRs).
__device__ __forceinline__ void async_load_b128(void* lds, const void* gptr) {
    asm volatile("global_load_async_to_lds_b128 %0, %1, off"
                 :
                 : "v"((unsigned)(unsigned long long)lds), "v"(gptr)
                 : "memory");
}
__device__ __forceinline__ void wait_asynccnt0() {
    asm volatile("s_wait_asynccnt 0x0" ::: "memory");
}

// ---------------------------------------------------------------------------
// Tiled bf16 WMMA GEMM:  out[M x 2048] = A[M x 2048] * W[2048 x 2048] + bias
// Block tile 128x128, K-tile 32, 256 threads = 8 waves, wave tile 64x32.
// Double-buffered LDS + software pipeline: stage next tile's global loads as
// raw registers during the WMMA phase; convert to bf16 only at commit time.
// ---------------------------------------------------------------------------
template <typename AT, bool OUT_BF16>
__global__ __launch_bounds__(256) void gemm_bf16_wmma(
    const AT* __restrict__ A,
    const float* __restrict__ W,
    const float* __restrict__ bias,
    void* __restrict__ outp,
    int M)
{
    (void)M;
    __shared__ alignas(16) unsigned short At[2][128 * 32];   // [m][k]
    __shared__ alignas(16) unsigned short Bt[2][128 * 32];   // transposed: [n][k]

    const int tid  = threadIdx.x;
    const int lane = tid & 31;
    const int wave = tid >> 5;
    const int wm   = wave >> 2;          // 0..1  (64-row stripe)
    const int wn   = wave & 3;           // 0..3  (32-col stripe)
    const int lrow = lane & 15;
    const int hi   = lane >> 4;          // 0/1: which 16-lane half

    const int mBlock = blockIdx.y * 128;
    const int nBlock = blockIdx.x * 128;

    const v8f z = {0.f,0.f,0.f,0.f,0.f,0.f,0.f,0.f};
    v8f acc[4][2];
#pragma unroll
    for (int mt = 0; mt < 4; ++mt)
#pragma unroll
        for (int nt = 0; nt < 2; ++nt) acc[mt][nt] = z;

    const int arow = tid >> 1;           // 0..127
    const int akh  = (tid & 1) * 16;     // 0/16
    const int bk   = tid >> 3;           // 0..31
    const int bnh  = (tid & 7) * 16;     // 0..112

    const int akOff = hi ? 8 : 0;        // A-frag chunk base (K 0..7 / 8..15)
    const int bkOff = hi ? 8 : 0;        // B-frag chunk base (K 0..7 / 8..15)

    // ---- staging registers (raw, converted only at commit) ----
    F4 saf[4];  // A tile chunk when AT==float
    U4 sah[2];  // A tile chunk when AT==bf16
    F4 sbf[4];  // B tile chunk (always fp32 weights)

    auto stage = [&](int kt) {
        if constexpr (sizeof(AT) == 4) {
            const float* src = (const float*)A + (size_t)(mBlock + arow) * HIDDEN + kt + akh;
            saf[0] = *(const F4*)(src + 0);
            saf[1] = *(const F4*)(src + 4);
            saf[2] = *(const F4*)(src + 8);
            saf[3] = *(const F4*)(src + 12);
        } else {
            const unsigned short* src = (const unsigned short*)A + (size_t)(mBlock + arow) * HIDDEN + kt + akh;
            sah[0] = *(const U4*)(src + 0);
            sah[1] = *(const U4*)(src + 8);
        }
        const float* bs = W + (size_t)(kt + bk) * HIDDEN + nBlock + bnh;
        sbf[0] = *(const F4*)(bs + 0);
        sbf[1] = *(const F4*)(bs + 4);
        sbf[2] = *(const F4*)(bs + 8);
        sbf[3] = *(const F4*)(bs + 12);
    };

    stage(0);

    for (int kt = 0; kt < HIDDEN; kt += 32) {
        const int buf = (kt >> 5) & 1;
        // ---- commit staged tile to LDS (convert fp32 -> bf16 here) ----
        {
            unsigned short tmp[16];
            if constexpr (sizeof(AT) == 4) {
                const float fs[16] = {saf[0].x,saf[0].y,saf[0].z,saf[0].w,
                                      saf[1].x,saf[1].y,saf[1].z,saf[1].w,
                                      saf[2].x,saf[2].y,saf[2].z,saf[2].w,
                                      saf[3].x,saf[3].y,saf[3].z,saf[3].w};
#pragma unroll
                for (int j = 0; j < 16; ++j) tmp[j] = f2bf(fs[j]);
            } else {
                *(U4*)&tmp[0] = sah[0];
                *(U4*)&tmp[8] = sah[1];
            }
            *(U4*)&At[buf][arow * 32 + akh + 0] = *(U4*)&tmp[0];
            *(U4*)&At[buf][arow * 32 + akh + 8] = *(U4*)&tmp[8];

            const float bsv[16] = {sbf[0].x,sbf[0].y,sbf[0].z,sbf[0].w,
                                   sbf[1].x,sbf[1].y,sbf[1].z,sbf[1].w,
                                   sbf[2].x,sbf[2].y,sbf[2].z,sbf[2].w,
                                   sbf[3].x,sbf[3].y,sbf[3].z,sbf[3].w};
#pragma unroll
            for (int j = 0; j < 16; ++j) Bt[buf][(bnh + j) * 32 + bk] = f2bf(bsv[j]);
        }
        __syncthreads();

        // ---- issue next tile's global loads (overlap with WMMA phase) ----
        if (kt + 32 < HIDDEN) stage(kt + 32);

        // ---- hoist all fragments, then 8 back-to-back WMMAs ----
        v16bf afr[4], bfr[2];
#pragma unroll
        for (int mt = 0; mt < 4; ++mt) {
            const unsigned short* base = &At[buf][(wm * 64 + mt * 16 + lrow) * 32];
            afr[mt] = frag2(base + akOff, base + akOff + 16);
        }
#pragma unroll
        for (int nt = 0; nt < 2; ++nt) {
            const unsigned short* base = &Bt[buf][(wn * 32 + nt * 16 + lrow) * 32];
            bfr[nt] = frag2(base + bkOff, base + bkOff + 16);
        }
#pragma unroll
        for (int nt = 0; nt < 2; ++nt)
#pragma unroll
            for (int mt = 0; mt < 4; ++mt)
                acc[mt][nt] = __builtin_amdgcn_wmma_f32_16x16x32_bf16(
                    false, afr[mt], false, bfr[nt], (short)0, acc[mt][nt], false, false);
    }

    // ---- epilogue: bias + store (C layout: VGPR i -> row i / i+8, lane -> col) ----
#pragma unroll
    for (int mt = 0; mt < 4; ++mt) {
#pragma unroll
        for (int nt = 0; nt < 2; ++nt) {
            const int col = nBlock + wn * 32 + nt * 16 + lrow;
            const float bv = bias[col];
#pragma unroll
            for (int i = 0; i < 8; ++i) {
                const int row = mBlock + wm * 64 + mt * 16 + i + hi * 8;
                const float v = acc[mt][nt][i] + bv;
                if constexpr (OUT_BF16)
                    ((unsigned short*)outp)[(size_t)row * HIDDEN + col] = f2bf(v);
                else
                    ((float*)outp)[(size_t)row * HIDDEN + col] = v;
            }
        }
    }
}

// ---------------------------------------------------------------------------
// Flash attention with WMMA. One workgroup = one (b, h, 128-q-row tile).
// 8 waves, each wave owns 16 q rows. K-tiles of 32 rows streamed through
// double-buffered LDS: K via async global->LDS copies (ASYNCcnt), V via
// registers (its LDS image is transposed). Single barrier per K-step.
// ---------------------------------------------------------------------------
__global__ __launch_bounds__(256) void flash_attn_wmma(
    const unsigned short* __restrict__ Q,
    const unsigned short* __restrict__ K,
    const unsigned short* __restrict__ V,
    unsigned short* __restrict__ Xo)
{
    __shared__ alignas(16) unsigned short Kt[2][32 * 128];    // [krow][d]
    __shared__ alignas(16) unsigned short Vt[2][128 * 32];    // transposed [d][krow]
    __shared__ alignas(16) unsigned short Pst[8][16 * 32];    // per-wave P staging [m][k]

    const int bid = blockIdx.x;
    const int qt  = bid & 15;
    const int h   = (bid >> 4) & 15;
    const int b   = bid >> 8;

    const int tid  = threadIdx.x;
    const int lane = tid & 31;
    const int wave = tid >> 5;
    const int lrow = lane & 15;
    const int hi   = lane >> 4;

    const float scale = 0.08838834764831845f;  // 1/sqrt(128)

    // ---- Q rows for this wave as 4 A-fragments (d split into 4 K=32 chunks) ----
    v16bf qfr[4];
    {
        const unsigned short* qp =
            Q + ((size_t)(b * SEQ + qt * 128 + wave * 16 + lrow)) * HIDDEN + h * HEAD_DIM;
        const int akOff = hi ? 8 : 0;
#pragma unroll
        for (int c = 0; c < 4; ++c)
            qfr[c] = frag2(qp + c * 32 + akOff, qp + c * 32 + akOff + 16);
    }

    float mrow[8], lsum[8], alpha[8];
#pragma unroll
    for (int r = 0; r < 8; ++r) { mrow[r] = -3.0e38f; lsum[r] = 0.f; }

    const v8f z = {0.f,0.f,0.f,0.f,0.f,0.f,0.f,0.f};
    v8f o[8];
#pragma unroll
    for (int dt = 0; dt < 8; ++dt) o[dt] = z;

    const int kr = tid >> 3;             // 0..31: k-row staged by this thread
    const int dh = (tid & 7) * 16;       // 0..112: d-chunk
    const size_t kvBase = (size_t)(b * SEQ) * HIDDEN + h * HEAD_DIM;

    union { U4 u[2]; unsigned short s[16]; } vv;   // V staging registers

    auto stage = [&](int kt, int buf) {
        const unsigned short* ksrc = K + kvBase + (size_t)(kt + kr) * HIDDEN + dh;
        async_load_b128(&Kt[buf][kr * 128 + dh + 0], ksrc + 0);
        async_load_b128(&Kt[buf][kr * 128 + dh + 8], ksrc + 8);
        const unsigned short* vsrc = V + kvBase + (size_t)(kt + kr) * HIDDEN + dh;
        vv.u[0] = *(const U4*)(vsrc + 0);
        vv.u[1] = *(const U4*)(vsrc + 8);
        if (kt + 32 < SEQ) __builtin_prefetch(vsrc + 32 * HIDDEN, 0, 0);
    };

    stage(0, 0);

    for (int kt = 0; kt < SEQ; kt += 32) {
        const int buf = (kt >> 5) & 1;
        // ---- commit transposed V tile; wait for this wave's async K copies ----
#pragma unroll
        for (int j = 0; j < 16; ++j) Vt[buf][(dh + j) * 32 + kr] = vv.s[j];
        wait_asynccnt0();
        __syncthreads();

        // ---- issue next tile (async K into other buffer, V into registers) ----
        if (kt + 32 < SEQ) stage(kt + 32, buf ^ 1);

        // ---- scores: hoist 4 K-fragments per 16-col tile, then 4 WMMAs ----
        v8f st[2] = {z, z};
        const int bkOff = hi ? 8 : 0;
#pragma unroll
        for (int t = 0; t < 2; ++t) {
            const unsigned short* kbase = &Kt[buf][(t * 16 + lrow) * 128];
            v16bf kf[4];
#pragma unroll
            for (int c = 0; c < 4; ++c)
                kf[c] = frag2(kbase + c * 32 + bkOff, kbase + c * 32 + bkOff + 16);
#pragma unroll
            for (int c = 0; c < 4; ++c)
                st[t] = __builtin_amdgcn_wmma_f32_16x16x32_bf16(
                    false, qfr[c], false, kf[c], (short)0, st[t], false, false);
        }

        // ---- online softmax (rows per-VGPR, cols across 16-lane half) ----
#pragma unroll
        for (int r = 0; r < 8; ++r) {
            float s0 = st[0][r] * scale, s1 = st[1][r] * scale;
            float mx = fmaxf(s0, s1);
            mx = fmaxf(mx, __shfl_xor(mx, 1, 32));
            mx = fmaxf(mx, __shfl_xor(mx, 2, 32));
            mx = fmaxf(mx, __shfl_xor(mx, 4, 32));
            mx = fmaxf(mx, __shfl_xor(mx, 8, 32));
            const float mNew = fmaxf(mrow[r], mx);
            alpha[r] = __expf(mrow[r] - mNew);
            const float p0 = __expf(s0 - mNew);
            const float p1 = __expf(s1 - mNew);
            float rs = p0 + p1;
            rs += __shfl_xor(rs, 1, 32);
            rs += __shfl_xor(rs, 2, 32);
            rs += __shfl_xor(rs, 4, 32);
            rs += __shfl_xor(rs, 8, 32);
            lsum[r] = lsum[r] * alpha[r] + rs;
            mrow[r] = mNew;
            st[0][r] = p0; st[1][r] = p1;
        }

        // ---- C->A relayout of P through per-wave LDS (DS in-order per wave) ----
        unsigned short* pst = &Pst[wave][0];
#pragma unroll
        for (int t = 0; t < 2; ++t)
#pragma unroll
            for (int r = 0; r < 8; ++r)
                pst[(r + hi * 8) * 32 + t * 16 + lrow] = f2bf(st[t][r]);
        v16bf pf;
        {
            const unsigned short* base = pst + lrow * 32 + (hi ? 8 : 0);
            pf = frag2(base, base + 16);
        }

        // ---- O = O*alpha + P @ V (pairs: 2 V-frag loads, then 2 WMMAs) ----
        const int vkOff = hi ? 8 : 0;
#pragma unroll
        for (int dt = 0; dt < 8; dt += 2) {
            const unsigned short* vb0 = &Vt[buf][((dt + 0) * 16 + lrow) * 32];
            const unsigned short* vb1 = &Vt[buf][((dt + 1) * 16 + lrow) * 32];
            v16bf vf0 = frag2(vb0 + vkOff, vb0 + vkOff + 16);
            v16bf vf1 = frag2(vb1 + vkOff, vb1 + vkOff + 16);
#pragma unroll
            for (int r = 0; r < 8; ++r) { o[dt][r] *= alpha[r]; o[dt + 1][r] *= alpha[r]; }
            o[dt]     = __builtin_amdgcn_wmma_f32_16x16x32_bf16(
                false, pf, false, vf0, (short)0, o[dt],     false, false);
            o[dt + 1] = __builtin_amdgcn_wmma_f32_16x16x32_bf16(
                false, pf, false, vf1, (short)0, o[dt + 1], false, false);
        }
    }

    // ---- normalize and scatter with the transpose(0,3,2,1)+reshape quirk ----
#pragma unroll
    for (int r = 0; r < 8; ++r) lsum[r] = 1.0f / lsum[r];
    const int sBase = qt * 128 + wave * 16 + hi * 8;
#pragma unroll
    for (int dt = 0; dt < 8; ++dt) {
        const int d = dt * 16 + lrow;
#pragma unroll
        for (int r = 0; r < 8; ++r) {
            const int s = sBase + r;
            const float v = o[dt][r] * lsum[r];
            const int rr = d * 16 + (s >> 7);          // row of reshaped matrix
            const int cc = (s & 127) * 16 + h;         // col of reshaped matrix
            Xo[((size_t)(b * SEQ + rr)) * HIDDEN + cc] = f2bf(v);
        }
    }
}

// ---------------------------------------------------------------------------
extern "C" void kernel_launch(void* const* d_in, const int* in_sizes, int n_in,
                              void* d_out, int out_size, void* d_ws, size_t ws_size,
                              hipStream_t stream) {
    (void)in_sizes; (void)n_in; (void)out_size; (void)ws_size;
    const float* X  = (const float*)d_in[0];
    const float* Wq = (const float*)d_in[1];
    const float* bq = (const float*)d_in[2];
    const float* Wk = (const float*)d_in[3];
    const float* bk = (const float*)d_in[4];
    const float* Wv = (const float*)d_in[5];
    const float* bv = (const float*)d_in[6];
    const float* Wo = (const float*)d_in[7];
    const float* bo = (const float*)d_in[8];
    float* out = (float*)d_out;

    unsigned short* ws = (unsigned short*)d_ws;
    const size_t MAT = (size_t)BATCH * SEQ * HIDDEN;   // 8M bf16 elements
    unsigned short* Qbf = ws + 0 * MAT;
    unsigned short* Kbf = ws + 1 * MAT;
    unsigned short* Vbf = ws + 2 * MAT;
    unsigned short* Xo  = ws + 3 * MAT;                // 64 MB total workspace

    const dim3 g(HIDDEN / 128, (BATCH * SEQ) / 128);   // (16, 32)
    gemm_bf16_wmma<float, true><<<g, 256, 0, stream>>>(X, Wq, bq, Qbf, BATCH * SEQ);
    gemm_bf16_wmma<float, true><<<g, 256, 0, stream>>>(X, Wk, bk, Kbf, BATCH * SEQ);
    gemm_bf16_wmma<float, true><<<g, 256, 0, stream>>>(X, Wv, bv, Vbf, BATCH * SEQ);
    flash_attn_wmma<<<BATCH * HEADS * (SEQ / 128), 256, 0, stream>>>(Qbf, Kbf, Vbf, Xo);
    gemm_bf16_wmma<unsigned short, false><<<g, 256, 0, stream>>>(Xo, Wo, bo, out, BATCH * SEQ);
}
